// SelfAttentionBlock_47708496724927
// MI455X (gfx1250) — matrix operand order repaired and checked by reading, not compiled, not run on hardware
//
#include <hip/hip_runtime.h>
#include <hip/hip_bf16.h>

#define BB 4
#define CC 256
#define NN 4096
#define DKK 32

typedef __attribute__((ext_vector_type(16))) __bf16 v16bf;
typedef __attribute__((ext_vector_type(8)))  float  v8f;
typedef __attribute__((ext_vector_type(4)))  float  v4f;
typedef __attribute__((ext_vector_type(4)))  unsigned int v4u;

union Vb { v16bf v; unsigned short u[16]; v4u q[2]; };

__device__ __forceinline__ unsigned short f2bf(float f) {
    unsigned int x = __builtin_bit_cast(unsigned int, f);
    unsigned int r = x + 0x7FFFu + ((x >> 16) & 1u);
    return (unsigned short)(r >> 16);
}

// Cooperative stage of 512 x b128 (128 rows x 32 bf16) global -> LDS using the
// CDNA5 async-to-LDS path (tracked by ASYNCcnt). Each wave waits for its own
// async ops before the caller's block barrier, so all data is visible after it.
__device__ __forceinline__ void stage_chunk_async(unsigned short* ldsf,
                                                  const unsigned short* src,
                                                  int tid)
{
    unsigned lbase = (unsigned)(unsigned long long)(uintptr_t)ldsf;
    #pragma unroll
    for (int p = 0; p < 2; ++p) {
        int idx = p * 256 + tid;                 // 0..511
        const unsigned short* g = src + idx * 8; // 8 bf16 = 16 bytes
        unsigned laddr = lbase + idx * 16;
        asm volatile("global_load_async_to_lds_b128 %0, %1, off"
                     :: "v"(laddr), "v"(g) : "memory");
    }
    asm volatile("s_wait_asynccnt 0" ::: "memory");
}

// ---------------------------------------------------------------------------
// Kernel 1: pointwise projections.
//   fT = (Wf x + bf)^T  bf16 [B][N][32]   (d contiguous -> B-operand loads)
//   gT = (Wg x + bg)^T  bf16 [B][N][32]   (d contiguous -> A-operand loads)
//   h  =  Wh x + bh     bf16 [B][C][N]    (j contiguous -> A-operand loads)
// ---------------------------------------------------------------------------
__global__ __launch_bounds__(256) void proj_kernel(
    const float* __restrict__ x,
    const float* __restrict__ Wf, const float* __restrict__ bf,
    const float* __restrict__ Wg, const float* __restrict__ bg,
    const float* __restrict__ Wh, const float* __restrict__ bh,
    unsigned short* __restrict__ fT, unsigned short* __restrict__ gT,
    unsigned short* __restrict__ hbf)
{
    __shared__ float xs[CC * 32];
    const int b   = blockIdx.y;
    const int n0  = blockIdx.x * 32;
    const int tid = threadIdx.x;
    const float* xb = x + (size_t)b * CC * NN;

    // stage x tile [256 c][32 n] : 2048 float4, 8 per thread
    #pragma unroll
    for (int p = 0; p < 8; ++p) {
        int idx = p * 256 + tid;
        int c = idx >> 3;
        int q = idx & 7;
        *(v4f*)(xs + c * 32 + q * 4) =
            *(const v4f*)(xb + (size_t)c * NN + n0 + q * 4);
    }
    __syncthreads();

    const int nl  = tid & 31;   // column within tile (lane)
    const int grp = tid >> 5;   // wave index: rows grp*40 .. grp*40+39
    const int n   = n0 + nl;

    for (int rr = 0; rr < 40; ++rr) {
        int r = grp * 40 + rr;
        const float* wrow;
        float bias;
        if (r < 32)      { wrow = Wf + r * CC;        bias = bf[r];       }
        else if (r < 64) { wrow = Wg + (r - 32) * CC; bias = bg[r - 32];  }
        else             { wrow = Wh + (r - 64) * CC; bias = bh[r - 64];  }
        float acc = bias;
        #pragma unroll 8
        for (int c = 0; c < CC; ++c)
            acc = fmaf(wrow[c], xs[c * 32 + nl], acc);
        unsigned short val = f2bf(acc);
        if (r < 32)
            fT[((size_t)b * NN + n) * DKK + r] = val;
        else if (r < 64)
            gT[((size_t)b * NN + n) * DKK + (r - 32)] = val;
        else
            hbf[(size_t)b * CC * NN + (size_t)(r - 64) * NN + n] = val;
    }
}

// ---------------------------------------------------------------------------
// Kernel 2: row stats of s = g^T f (softmax over i, per row j). Two phases:
//   A) per-lane running max over i, one butterfly at the end -> m_j
//   B) per-lane running sum of exp(s-m_j), one butterfly -> L_j
// One wave owns a 16-row j-tile; f staged in LDS chunks of 128 i (async).
// ---------------------------------------------------------------------------
__global__ __launch_bounds__(256) void rowstats_kernel(
    const unsigned short* __restrict__ fT,
    const unsigned short* __restrict__ gT,
    float* __restrict__ wm, float* __restrict__ wr)
{
    __shared__ unsigned short ldsf[128 * DKK];   // f chunk [128 i][32 d]
    const int b    = blockIdx.y;
    const int tid  = threadIdx.x;
    const int w    = tid >> 5;
    const int l    = tid & 31;
    const int half = l >> 4;
    const int li   = l & 15;
    const unsigned short* fb = fT + (size_t)b * NN * DKK;
    const unsigned short* gb = gT + (size_t)b * NN * DKK;
    const int j0 = blockIdx.x * 128 + w * 16;

    // A operand: g^T tile [16 j][32 d]; K runs are contiguous in gT
    Vb a;
    {
        const unsigned short* gr = gb + (size_t)(j0 + li) * DKK;
        a.q[0] = *(const v4u*)(gr + 16 * half);
        a.q[1] = *(const v4u*)(gr + 16 * half + 8);
    }

    const v8f czero = {0.f, 0.f, 0.f, 0.f, 0.f, 0.f, 0.f, 0.f};

    // ---- phase A: row maxima ----
    float mrow[8];
    #pragma unroll
    for (int v = 0; v < 8; ++v) mrow[v] = -3.0e38f;

    for (int ic = 0; ic < NN / 128; ++ic) {
        __syncthreads();
        stage_chunk_async(ldsf, fb + (size_t)ic * 128 * DKK, tid);
        __syncthreads();
        #pragma unroll
        for (int it = 0; it < 8; ++it) {
            Vb bt;
            const unsigned short* fr = ldsf + (it * 16 + li) * DKK + 16 * half;
            bt.q[0] = *(const v4u*)(fr);
            bt.q[1] = *(const v4u*)(fr + 8);
            v8f s = __builtin_amdgcn_wmma_f32_16x16x32_bf16(
                false, a.v, false, bt.v, (short)0, czero, false, false);
            #pragma unroll
            for (int v = 0; v < 8; ++v) mrow[v] = fmaxf(mrow[v], s[v]);
        }
    }
    #pragma unroll
    for (int v = 0; v < 8; ++v)
        #pragma unroll
        for (int msk = 1; msk < 16; msk <<= 1)
            mrow[v] = fmaxf(mrow[v], __shfl_xor(mrow[v], msk, 32));

    // ---- phase B: row sums of exp ----
    float psum[8];
    #pragma unroll
    for (int v = 0; v < 8; ++v) psum[v] = 0.0f;

    for (int ic = 0; ic < NN / 128; ++ic) {
        __syncthreads();
        stage_chunk_async(ldsf, fb + (size_t)ic * 128 * DKK, tid);
        __syncthreads();
        #pragma unroll
        for (int it = 0; it < 8; ++it) {
            Vb bt;
            const unsigned short* fr = ldsf + (it * 16 + li) * DKK + 16 * half;
            bt.q[0] = *(const v4u*)(fr);
            bt.q[1] = *(const v4u*)(fr + 8);
            v8f s = __builtin_amdgcn_wmma_f32_16x16x32_bf16(
                false, a.v, false, bt.v, (short)0, czero, false, false);
            #pragma unroll
            for (int v = 0; v < 8; ++v) psum[v] += __expf(s[v] - mrow[v]);
        }
    }
    #pragma unroll
    for (int v = 0; v < 8; ++v)
        #pragma unroll
        for (int msk = 1; msk < 16; msk <<= 1)
            psum[v] += __shfl_xor(psum[v], msk, 32);

    if (li == 0) {
        #pragma unroll
        for (int v = 0; v < 8; ++v) {
            int j = j0 + half * 8 + v;
            wm[(size_t)b * NN + j] = mrow[v];
            wr[(size_t)b * NN + j] = 1.0f / psum[v];
        }
    }
}

// ---------------------------------------------------------------------------
// Kernel 3: O[c,i] = sum_j h[c,j] * exp(s[j,i]-m_j)/L_j ; out = O + x.
// Block: 64 i-columns x full C for one batch. Wave: one 16-col i-tile and
// a 128-row c-half (8 accumulator tiles). P transposed via per-wave LDS.
// ---------------------------------------------------------------------------
__global__ __launch_bounds__(256) void attnout_kernel(
    const unsigned short* __restrict__ fT,
    const unsigned short* __restrict__ gT,
    const unsigned short* __restrict__ hbf,
    const float* __restrict__ wm, const float* __restrict__ wr,
    const float* __restrict__ x, float* __restrict__ out)
{
    __shared__ float ldsm[NN];
    __shared__ float ldsr[NN];
    __shared__ float sbuf[8][32 * 16];   // per wave: s chunk [32 j][16 i]

    const int b    = blockIdx.y;
    const int tid  = threadIdx.x;
    const int w    = tid >> 5;
    const int l    = tid & 31;
    const int half = l >> 4;
    const int li   = l & 15;

    // stage m and 1/L for the whole batch (32 KB)
    for (int idx = tid; idx < NN / 4; idx += 256) {
        ((v4f*)ldsm)[idx] = ((const v4f*)(wm + (size_t)b * NN))[idx];
        ((v4f*)ldsr)[idx] = ((const v4f*)(wr + (size_t)b * NN))[idx];
    }
    __syncthreads();

    const int i0     = blockIdx.x * 64 + (w >> 1) * 16;
    const int c0base = (w & 1) * 128;
    const unsigned short* fb = fT  + (size_t)b * NN * DKK;
    const unsigned short* gb = gT  + (size_t)b * NN * DKK;
    const unsigned short* hb = hbf + (size_t)b * CC * NN;

    // fixed B operand: f tile [32 d][16 i] for this wave's i-tile
    Vb btf;
    {
        const unsigned short* fr = fb + (size_t)(i0 + li) * DKK + 16 * half;
        btf.q[0] = *(const v4u*)(fr);
        btf.q[1] = *(const v4u*)(fr + 8);
    }

    v8f acc[8];
    #pragma unroll
    for (int t = 0; t < 8; ++t)
        acc[t] = (v8f){0.f, 0.f, 0.f, 0.f, 0.f, 0.f, 0.f, 0.f};
    const v8f czero = {0.f, 0.f, 0.f, 0.f, 0.f, 0.f, 0.f, 0.f};

    float* sb = sbuf[w];

    for (int jc = 0; jc < NN; jc += 32) {
        // A operands: g^T tiles rows jc..jc+15 and jc+16..jc+31
        Vb a0, a1;
        {
            const unsigned short* g0 = gb + (size_t)(jc + li) * DKK;
            const unsigned short* g1 = gb + (size_t)(jc + 16 + li) * DKK;
            a0.q[0] = *(const v4u*)(g0 + 16 * half);
            a0.q[1] = *(const v4u*)(g0 + 16 * half + 8);
            a1.q[0] = *(const v4u*)(g1 + 16 * half);
            a1.q[1] = *(const v4u*)(g1 + 16 * half + 8);
        }
        v8f s0 = __builtin_amdgcn_wmma_f32_16x16x32_bf16(
            false, a0.v, false, btf.v, (short)0, czero, false, false);
        v8f s1 = __builtin_amdgcn_wmma_f32_16x16x32_bf16(
            false, a1.v, false, btf.v, (short)0, czero, false, false);

        // transpose C/D layout -> B-operand layout via per-wave LDS
        #pragma unroll
        for (int v = 0; v < 8; ++v) {
            sb[(v + 8 * half) * 16 + li]      = s0[v];
            sb[(16 + v + 8 * half) * 16 + li] = s1[v];
        }
        asm volatile("s_wait_dscnt 0" ::: "memory");
        __builtin_amdgcn_wave_barrier();

        // P tile as B operand: element e -> j = jc + 16*half + e
        Vb p;
        #pragma unroll
        for (int e = 0; e < 16; ++e) {
            int jl = half * 16 + e;
            int j  = jc + jl;
            float sv = sb[jl * 16 + li];
            float pv = __expf(sv - ldsm[j]) * ldsr[j];
            p.u[e] = f2bf(pv);
        }
        __builtin_amdgcn_wave_barrier();

        // accumulate: O_tile += H_tile(16c x 32j) * P(32j x 16i)
        #pragma unroll
        for (int ct = 0; ct < 8; ++ct) {
            int c0 = c0base + ct * 16;
            Vb ah;
            const unsigned short* hr = hb + (size_t)(c0 + li) * NN + jc;
            ah.q[0] = *(const v4u*)(hr + 16 * half);
            ah.q[1] = *(const v4u*)(hr + 16 * half + 8);
            acc[ct] = __builtin_amdgcn_wmma_f32_16x16x32_bf16(
                false, ah.v, false, p.v, (short)0, acc[ct], false, false);
        }
    }

    // epilogue: residual add + store
    const float* xb = x + (size_t)b * CC * NN;
    float* ob = out + (size_t)b * CC * NN;
    const int i = i0 + li;
    #pragma unroll
    for (int ct = 0; ct < 8; ++ct) {
        int c0 = c0base + ct * 16;
        #pragma unroll
        for (int v = 0; v < 8; ++v) {
            int c = c0 + v + 8 * half;
            ob[(size_t)c * NN + i] = acc[ct][v] + xb[(size_t)c * NN + i];
        }
    }
}

// ---------------------------------------------------------------------------
extern "C" void kernel_launch(void* const* d_in, const int* in_sizes, int n_in,
                              void* d_out, int out_size, void* d_ws, size_t ws_size,
                              hipStream_t stream)
{
    (void)in_sizes; (void)n_in; (void)out_size; (void)ws_size;
    const float* x  = (const float*)d_in[0];
    const float* Wf = (const float*)d_in[1];
    const float* bf = (const float*)d_in[2];
    const float* Wg = (const float*)d_in[3];
    const float* bg = (const float*)d_in[4];
    const float* Wh = (const float*)d_in[5];
    const float* bh = (const float*)d_in[6];
    float* out = (float*)d_out;

    char* ws = (char*)d_ws;
    // ws layout (bytes):
    //   [0, 1MB)        fT bf16 [B][4096][32]
    //   [1MB, 2MB)      gT bf16 [B][4096][32]
    //   [2MB, 10MB)     h  bf16 [B][256][4096]
    //   [10MB, +64KB)   rowmax   f32 [B][4096]
    //   [.., +64KB)     1/rowsum f32 [B][4096]
    unsigned short* fT  = (unsigned short*)(ws);
    unsigned short* gT  = (unsigned short*)(ws + (size_t)(1 << 20));
    unsigned short* hbf = (unsigned short*)(ws + (size_t)(2 << 20));
    float* wm = (float*)(ws + (size_t)10485760);
    float* wr = (float*)(ws + (size_t)10485760 + 65536);

    proj_kernel<<<dim3(NN / 32, BB), 256, 0, stream>>>(
        x, Wf, bf, Wg, bg, Wh, bh, fT, gT, hbf);
    rowstats_kernel<<<dim3(NN / 128, BB), 256, 0, stream>>>(fT, gT, wm, wr);
    attnout_kernel<<<dim3(NN / 64, BB), 256, 0, stream>>>(
        fT, gT, hbf, wm, wr, x, out);
}
